// DGCNN_type4_87076166959375
// MI455X (gfx1250) — compile-verified, matrix operand order checked by reading, not run on hardware
//
#include <hip/hip_runtime.h>

typedef __attribute__((ext_vector_type(16))) _Float16 v16h;
typedef __attribute__((ext_vector_type(8)))  _Float16 v8h;
typedef __attribute__((ext_vector_type(8)))  float    v8f;

#define BATCH   16
#define NPTS    2048
#define KNN     16
#define BN      (BATCH*NPTS)
#define FSTRIDE 32
#define COMBW   197
#define LIN1KP  224   // 197 padded to multiple of 32

__device__ __forceinline__ float lrelu_f(float v){ return v > 0.f ? v : 0.01f*v; }

// order-preserving float <-> uint mapping for atomicMax pooling
__device__ __forceinline__ unsigned encf(float f){
  unsigned u = __float_as_uint(f);
  return (u & 0x80000000u) ? ~u : (u | 0x80000000u);
}
__device__ __forceinline__ float decf(unsigned u){
  unsigned b = (u & 0x80000000u) ? (u & 0x7FFFFFFFu) : ~u;
  return __uint_as_float(b);
}

// Load one lane's v16h operand fragment for a 16x32 f16 tile at `p`
// (p already offset to this lane's row + K-group base). Chunk0 -> VGPR0..3
// (K = kb..kb+7), chunk1 at +16 halves -> VGPR4..7 (K = 16+kb..).
__device__ __forceinline__ v16h load_half_tile(const _Float16* p){
  v8h lo = *(const v8h*)p;
  v8h hi = *(const v8h*)(p + 16);
  return __builtin_shufflevector(lo, hi, 0,1,2,3,4,5,6,7,8,9,10,11,12,13,14,15);
}

// ---------------------------------------------------------------- weight prep
// W [fi x fo] fp32 -> Wt [fo x fiPad] f16 (transposed, K zero-padded)
__global__ void wprep_kernel(const float* __restrict__ W, _Float16* __restrict__ Wt,
                             int fi, int fo, int fiPad){
  int t = blockIdx.x*blockDim.x + threadIdx.x;
  if (t >= fo*fiPad) return;
  int n = t / fiPad, k = t - n*fiPad;
  Wt[t] = (_Float16)(k < fi ? W[(size_t)k*fo + n] : 0.f);
}

// ---------------------------------------------------------------- input build
__global__ void build_xx_kernel(const float* __restrict__ x, const float* __restrict__ pos,
                                float* __restrict__ comb,
                                float* __restrict__ f1, float* __restrict__ f2){
  int p = blockIdx.x*blockDim.x + threadIdx.x;
  if (p >= BN) return;
  float xv = x[p];
  float p0 = pos[p*3+0], p1 = pos[p*3+1], p2 = pos[p*3+2];
  comb[(size_t)p*COMBW + 0]   = p0;
  comb[(size_t)p*COMBW + 1]   = p1;
  comb[(size_t)p*COMBW + 2]   = xv;
  comb[(size_t)p*COMBW + 99]  = p2;
  comb[(size_t)p*COMBW + 100] = xv;
#pragma unroll
  for (int f = 0; f < FSTRIDE; ++f){ f1[p*FSTRIDE+f] = 0.f; f2[p*FSTRIDE+f] = 0.f; }
  f1[p*FSTRIDE+0] = p0; f1[p*FSTRIDE+1] = p1; f1[p*FSTRIDE+2] = xv;
  f2[p*FSTRIDE+0] = p2; f2[p*FSTRIDE+1] = xv;
}

// ---------------------------------------------------------------- kNN (top-16)
// feat rows are zero-padded to 32 dims, so a fixed 32-dim distance is exact.
__global__ void knn_kernel(const float* __restrict__ feat, int* __restrict__ idx){
  __shared__ float tile[128*FSTRIDE];
  int tid = threadIdx.x;
  int p = blockIdx.x*256 + tid;           // all 256 pts in a block share a batch
  int b = p >> 11;                        // N = 2048
  float xi[FSTRIDE];
#pragma unroll
  for (int f = 0; f < FSTRIDE; ++f) xi[f] = feat[p*FSTRIDE + f];
  float bd[KNN]; int bi[KNN];
#pragma unroll
  for (int t = 0; t < KNN; ++t){ bd[t] = 3.0e38f; bi[t] = p; }
  for (int tb = 0; tb < NPTS; tb += 128){
    __syncthreads();
    for (int i = tid; i < 128*FSTRIDE; i += 256)
      tile[i] = feat[((size_t)b*NPTS + tb)*FSTRIDE + i];
    __syncthreads();
    for (int c = 0; c < 128; ++c){
      float d2 = 0.f;
#pragma unroll
      for (int f = 0; f < FSTRIDE; ++f){
        float df = xi[f] - tile[c*FSTRIDE + f];
        d2 += df*df;
      }
      if (d2 < bd[KNN-1]){
        bd[KNN-1] = d2; bi[KNN-1] = b*NPTS + tb + c;
#pragma unroll
        for (int t = KNN-1; t > 0; --t){
          if (bd[t] < bd[t-1]){
            float td = bd[t]; bd[t] = bd[t-1]; bd[t-1] = td;
            int ti = bi[t]; bi[t] = bi[t-1]; bi[t-1] = ti;
          }
        }
      }
    }
  }
#pragma unroll
  for (int t = 0; t < KNN; ++t) idx[p*KNN + t] = bi[t];
}

// ------------------------------------------------- fused edge-conv (WMMA x2)
// One wave = one point = one 16-edge M-tile. e=[xi, xj-xi] -> W1 -> lrelu ->
// W2 -> lrelu -> max over the 16 edges. Weights staged once per workgroup.
template<int D, int KPAD, int H1, int H2>
__global__ void edgeconv_kernel(const float* __restrict__ feat,
                                const int* __restrict__ idx,
                                const _Float16* __restrict__ Wt1, const float* __restrict__ B1,
                                const _Float16* __restrict__ Wt2, const float* __restrict__ B2,
                                float* __restrict__ comb, int colBase,
                                float* __restrict__ featOut){
  __shared__ __attribute__((aligned(16))) _Float16 sW1[H1*KPAD];
  __shared__ __attribute__((aligned(16))) _Float16 sW2[H2*H1];
  __shared__ __attribute__((aligned(16))) _Float16 sEdge[4*16*KPAD];
  __shared__ __attribute__((aligned(16))) _Float16 sHid[4*16*H1];

  int tid  = threadIdx.x;
  int wv   = tid >> 5;
  int lane = tid & 31;
  int p    = blockIdx.x*4 + wv;

  for (int i = tid; i < H1*KPAD; i += 128) sW1[i] = Wt1[i];
  for (int i = tid; i < H2*H1;  i += 128) sW2[i] = Wt2[i];
  __syncthreads();

  _Float16* eT = sEdge + wv*16*KPAD;
  _Float16* hT = sHid  + wv*16*H1;

  // build the 16 x KPAD edge tile: [xi(D), xj-xi(D), zero pad]
  for (int e = lane; e < 16*KPAD; e += 32){
    int row = e / KPAD;
    int c   = e - row*KPAD;
    float v = 0.f;
    if (c < D){
      v = feat[p*FSTRIDE + c];
    } else if (c < 2*D){
      int nbr = idx[p*KNN + row];
      int f   = c - D;
      v = feat[nbr*FSTRIDE + f] - feat[p*FSTRIDE + f];
    }
    eT[e] = (_Float16)v;
  }
  __syncthreads();

  int m16 = lane & 15;
  int kb  = (lane >> 4) << 3;
  int mhi = (lane >> 4) << 3;

  // layer 1: 16xKPAD @ KPADxH1 -> lrelu -> LDS (f16)
#pragma unroll
  for (int nt = 0; nt < H1/16; ++nt){
    v8f acc = {};
#pragma unroll
    for (int kc = 0; kc < KPAD/32; ++kc){
      v16h a  = load_half_tile(eT  + m16*KPAD + kc*32 + kb);
      v16h bm = load_half_tile(sW1 + (nt*16 + m16)*KPAD + kc*32 + kb);
      acc = __builtin_amdgcn_wmma_f32_16x16x32_f16(false, a, false, bm,
                                                   (short)0, acc, false, false);
    }
    float bv = B1[nt*16 + m16];
#pragma unroll
    for (int r = 0; r < 8; ++r){
      float v = lrelu_f(acc[r] + bv);
      hT[(r + mhi)*H1 + nt*16 + m16] = (_Float16)v;
    }
  }
  __syncthreads();

  // layer 2: 16xH1 @ H1xH2 -> lrelu -> max over 16 edge rows
#pragma unroll
  for (int nt = 0; nt < H2/16; ++nt){
    v8f acc = {};
#pragma unroll
    for (int kc = 0; kc < H1/32; ++kc){
      v16h a  = load_half_tile(hT  + m16*H1 + kc*32 + kb);
      v16h bm = load_half_tile(sW2 + (nt*16 + m16)*H1 + kc*32 + kb);
      acc = __builtin_amdgcn_wmma_f32_16x16x32_f16(false, a, false, bm,
                                                   (short)0, acc, false, false);
    }
    float bv = B2[nt*16 + m16];
    float vmax = -3.0e38f;
#pragma unroll
    for (int r = 0; r < 8; ++r)
      vmax = fmaxf(vmax, lrelu_f(acc[r] + bv));
    vmax = fmaxf(vmax, __shfl_xor(vmax, 16, 32));  // combine lane halves (M 0-7 / 8-15)
    if (lane < 16){
      int col = nt*16 + lane;
      comb[(size_t)p*COMBW + colBase + col] = vmax;
      featOut[p*FSTRIDE + col] = vmax;
    }
  }
}

// ------------------------------------------------------------- comb -> f16 A
__global__ void pack_kernel(const float* __restrict__ comb, _Float16* __restrict__ A){
  long t = (long)blockIdx.x*blockDim.x + threadIdx.x;
  if (t >= (long)BN*LIN1KP) return;
  int row = (int)(t / LIN1KP), c = (int)(t - (long)row*LIN1KP);
  A[t] = (_Float16)(c < COMBW ? comb[(size_t)row*COMBW + c] : 0.f);
}

__global__ void initpool_kernel(unsigned* __restrict__ pool){
  int t = blockIdx.x*blockDim.x + threadIdx.x;
  if (t < BATCH*256) pool[t] = 0u;
}

// ---------------------------------------------------- WMMA GEMM (lin1 layers)
// C = A[MxK] * Bt^T[KxN] + bias; mode 0: store f16 (optional lrelu);
// mode 1: per-column max over 16-row tile + atomicMax pooling per batch.
__global__ void gemm_kernel(const _Float16* __restrict__ A, int K,
                            const _Float16* __restrict__ Bt,
                            const float* __restrict__ bias,
                            int N, int doLrelu, int mode,
                            _Float16* __restrict__ outH,
                            unsigned* __restrict__ poolEnc){
  int lane  = threadIdx.x & 31;
  int wave  = (int)(((long)blockIdx.x*blockDim.x + threadIdx.x) >> 5);
  int ntile = N >> 4;
  int mt    = wave / ntile;
  int nt    = wave - mt*ntile;
  int row0  = mt << 4;
  int m16   = lane & 15;
  int kb    = (lane >> 4) << 3;
  int mhi   = (lane >> 4) << 3;

  v8f acc = {};
  const _Float16* aBase = A  + (size_t)(row0 + m16)*K + kb;
  const _Float16* bBase = Bt + (size_t)(nt*16 + m16)*K + kb;
  for (int kc = 0; kc < K; kc += 32){
    v16h a  = load_half_tile(aBase + kc);
    v16h bm = load_half_tile(bBase + kc);
    acc = __builtin_amdgcn_wmma_f32_16x16x32_f16(false, a, false, bm,
                                                 (short)0, acc, false, false);
  }
  float bv = bias[nt*16 + m16];
  if (mode == 0){
#pragma unroll
    for (int r = 0; r < 8; ++r){
      float v = acc[r] + bv;
      if (doLrelu) v = lrelu_f(v);
      outH[(size_t)(row0 + r + mhi)*N + nt*16 + m16] = (_Float16)v;
    }
  } else {
    float vmax = -3.0e38f;
#pragma unroll
    for (int r = 0; r < 8; ++r){
      float v = acc[r] + bv;
      if (doLrelu) v = lrelu_f(v);
      vmax = fmaxf(vmax, v);
    }
    vmax = fmaxf(vmax, __shfl_xor(vmax, 16, 32));
    if (lane < 16){
      int bidx = row0 >> 11;   // 16 rows always within one batch (2048 % 16 == 0)
      atomicMax(&poolEnc[bidx*N + nt*16 + lane], encf(vmax));
    }
  }
}

// ------------------------------------------------------------------- head MLP
__global__ void head_kernel(const unsigned* __restrict__ pool,
                            const float* __restrict__ wm1, const float* __restrict__ bm1,
                            const float* __restrict__ wm2, const float* __restrict__ bm2,
                            float* __restrict__ out){
  __shared__ float h[128];
  int b = blockIdx.x, j = threadIdx.x;
  float acc = bm1[j];
  for (int i = 0; i < 256; ++i){
    float pv = lrelu_f(decf(pool[b*256 + i]));
    acc += pv * wm1[i*128 + j];
  }
  h[j] = lrelu_f(acc);
  __syncthreads();
  if (j < 3){
    float o = bm2[j];
    for (int t = 0; t < 128; ++t) o += h[t]*wm2[t*3 + j];
    out[b*3 + j] = o;
  }
}

// =============================================================== launch logic
extern "C" void kernel_launch(void* const* d_in, const int* in_sizes, int n_in,
                              void* d_out, int out_size, void* d_ws, size_t ws_size,
                              hipStream_t stream){
  (void)in_sizes; (void)n_in; (void)out_size; (void)ws_size;
  const float* x    = (const float*)d_in[0];
  const float* pos  = (const float*)d_in[1];
  // d_in[2] = tq (unused, FEA==1), d_in[3] = batch (implicit layout)
  const float* c11w1=(const float*)d_in[4];  const float* c11b1=(const float*)d_in[5];
  const float* c11w2=(const float*)d_in[6];  const float* c11b2=(const float*)d_in[7];
  const float* c12w1=(const float*)d_in[8];  const float* c12b1=(const float*)d_in[9];
  const float* c12w2=(const float*)d_in[10]; const float* c12b2=(const float*)d_in[11];
  const float* c2w1 =(const float*)d_in[12]; const float* c2b1 =(const float*)d_in[13];
  const float* c2w2 =(const float*)d_in[14]; const float* c2b2 =(const float*)d_in[15];
  const float* l1w1 =(const float*)d_in[16]; const float* l1b1 =(const float*)d_in[17];
  const float* l1w2 =(const float*)d_in[18]; const float* l1b2 =(const float*)d_in[19];
  const float* mw1  =(const float*)d_in[20]; const float* mb1  =(const float*)d_in[21];
  const float* mw2  =(const float*)d_in[22]; const float* mb2  =(const float*)d_in[23];
  float* out = (float*)d_out;

  char* wsp = (char*)d_ws; size_t off = 0;
  auto carve = [&](size_t bytes)->char*{
    char* pp = wsp + off; off += (bytes + 255) & ~(size_t)255; return pp;
  };
  float*    comb  = (float*)   carve((size_t)BN*COMBW*4);
  _Float16* Alin  = (_Float16*)carve((size_t)BN*LIN1KP*2);
  _Float16* Hbuf  = (_Float16*)carve((size_t)BN*512*2);
  float*    f1    = (float*)   carve((size_t)BN*FSTRIDE*4);
  float*    f2    = (float*)   carve((size_t)BN*FSTRIDE*4);
  float*    g1    = (float*)   carve((size_t)BN*FSTRIDE*4);
  float*    g2    = (float*)   carve((size_t)BN*FSTRIDE*4);
  float*    q1    = (float*)   carve((size_t)BN*FSTRIDE*4);
  float*    q2    = (float*)   carve((size_t)BN*FSTRIDE*4);
  int*      idx   = (int*)     carve((size_t)BN*KNN*4);
  unsigned* pool  = (unsigned*)carve(BATCH*256*4);
  _Float16* tw11a = (_Float16*)carve(32*32*2);
  _Float16* tw11b = (_Float16*)carve(32*32*2);
  _Float16* tw12a = (_Float16*)carve(32*32*2);
  _Float16* tw12b = (_Float16*)carve(32*32*2);
  _Float16* tw2a  = (_Float16*)carve(64*64*2);
  _Float16* tw2b  = (_Float16*)carve(32*64*2);
  _Float16* twl1a = (_Float16*)carve((size_t)512*LIN1KP*2);
  _Float16* twl1b = (_Float16*)carve((size_t)256*512*2);

  auto wgrid = [](int n){ return (n + 255)/256; };
  // weight prep: W[fi x fo] -> Wt[fo x fiPad] f16
  wprep_kernel<<<wgrid(32*32), 256, 0, stream>>>(c11w1, tw11a,   6,  32,  32);
  wprep_kernel<<<wgrid(32*32), 256, 0, stream>>>(c11w2, tw11b,  32,  32,  32);
  wprep_kernel<<<wgrid(32*32), 256, 0, stream>>>(c12w1, tw12a,   4,  32,  32);
  wprep_kernel<<<wgrid(32*32), 256, 0, stream>>>(c12w2, tw12b,  32,  32,  32);
  wprep_kernel<<<wgrid(64*64), 256, 0, stream>>>(c2w1,  tw2a,   64,  64,  64);
  wprep_kernel<<<wgrid(32*64), 256, 0, stream>>>(c2w2,  tw2b,   64,  32,  64);
  wprep_kernel<<<wgrid(512*LIN1KP), 256, 0, stream>>>(l1w1, twl1a, 197, 512, LIN1KP);
  wprep_kernel<<<wgrid(256*512),    256, 0, stream>>>(l1w2, twl1b, 512, 256, 512);

  build_xx_kernel<<<BN/256, 256, 0, stream>>>(x, pos, comb, f1, f2);

  // depth 1
  knn_kernel<<<BN/256, 256, 0, stream>>>(f1, idx);
  edgeconv_kernel<3,32,32,32><<<BN/4, 128, 0, stream>>>(f1, idx, tw11a, c11b1, tw11b, c11b2, comb, 3,   g1);
  knn_kernel<<<BN/256, 256, 0, stream>>>(f2, idx);
  edgeconv_kernel<2,32,32,32><<<BN/4, 128, 0, stream>>>(f2, idx, tw12a, c12b1, tw12b, c12b2, comb, 101, g2);
  // depth 2 (shared conv2 weights)
  knn_kernel<<<BN/256, 256, 0, stream>>>(g1, idx);
  edgeconv_kernel<32,64,64,32><<<BN/4, 128, 0, stream>>>(g1, idx, tw2a, c2b1, tw2b, c2b2, comb, 35,  q1);
  knn_kernel<<<BN/256, 256, 0, stream>>>(g2, idx);
  edgeconv_kernel<32,64,64,32><<<BN/4, 128, 0, stream>>>(g2, idx, tw2a, c2b1, tw2b, c2b2, comb, 133, q2);
  // depth 3 (featOut dumped into f1, unused downstream)
  knn_kernel<<<BN/256, 256, 0, stream>>>(q1, idx);
  edgeconv_kernel<32,64,64,32><<<BN/4, 128, 0, stream>>>(q1, idx, tw2a, c2b1, tw2b, c2b2, comb, 67,  f1);
  knn_kernel<<<BN/256, 256, 0, stream>>>(q2, idx);
  edgeconv_kernel<32,64,64,32><<<BN/4, 128, 0, stream>>>(q2, idx, tw2a, c2b1, tw2b, c2b2, comb, 165, f1);

  // lin1: (BN x 197) -> lrelu(512) -> 256, fused global max-pool
  pack_kernel<<<(int)(((long)BN*LIN1KP + 255)/256), 256, 0, stream>>>(comb, Alin);
  initpool_kernel<<<wgrid(BATCH*256), 256, 0, stream>>>(pool);
  { // layer 1: M=32768 K=224 N=512 -> f16, lrelu
    int waves = (BN/16) * (512/16);
    gemm_kernel<<<waves/4, 128, 0, stream>>>(Alin, LIN1KP, twl1a, l1b1, 512, 1, 0, Hbuf, nullptr);
  }
  { // layer 2: M=32768 K=512 N=256 -> pooled max (no lrelu)
    int waves = (BN/16) * (256/16);
    gemm_kernel<<<waves/4, 128, 0, stream>>>(Hbuf, 512, twl1b, l1b2, 256, 0, 1, nullptr, pool);
  }

  head_kernel<<<BATCH, 128, 0, stream>>>(pool, mw1, mb1, mw2, mb2, out);
}
